// ODERNN_67061619360409
// MI455X (gfx1250) — compile-verified, matrix operand order checked by reading, not run on hardware
//
#include <hip/hip_runtime.h>
#include <hip/hip_bf16.h>
#include <math.h>

// ---------------------------------------------------------------------------
// ODE-RNN scan for MI455X (gfx1250, wave32, WMMA).
//  Pass 0: pack weights (f16 WMMA fragments, transposes, W2H = w2@w_hh fold).
//  Pass 1: 16 persistent workgroups; each wave pins its w_hh B-fragments in
//          VGPRs for the whole scan; one global barrier per step; h@w_hh via
//          two independent v_wmma_f32_16x16x32_f16 accumulation chains.
//  Pass 2: heads (relu-linear + mu / softplus-sigma) as parallel WMMA GEMMs.
// ---------------------------------------------------------------------------

typedef __attribute__((ext_vector_type(16))) _Float16 v16h;
typedef __attribute__((ext_vector_type(8)))  float    v8f;

#define T_STEPS 16384
#define IN_DIM  64
#define HID     1024
#define ODE_H   50
#define HHALF   512
#define NWG     16        // workgroups in pass-1 scan (one per WGP)
#define TPB     256       // 8 wave32 per workgroup
#define COLS    64        // HID / NWG columns owned per WGP
#define KB_HID  32        // 1024 / 32 K-blocks

// ---- workspace layout (bytes, 256-aligned regions) ----
#define OFF_BAR    ((size_t)0)                        // T ints (per-step barrier)
#define OFF_W2HT   (OFF_BAR   + (size_t)T_STEPS*4)    // [1024][64] f32 (W2H^T, j<50 valid)
#define OFF_B2H    (OFF_W2HT  + (size_t)HID*64*4)     // [1024] f32  (b2 @ w_hh)
#define OFF_BSUM   (OFF_B2H   + (size_t)HID*4)        // [1024] f32  (b_ih + b_hh)
#define OFF_W1T    (OFF_BSUM  + (size_t)HID*4)        // [50][1024] f16 (w1^T)
#define OFF_WIHT   (OFF_W1T   + (size_t)ODE_H*HID*2)  // [1024][64] f32 (w_ih^T)
#define OFF_WHHF   (OFF_WIHT  + (size_t)HID*IN_DIM*4) // w_hh f16 frags: 64nb*32kb*512
#define OFF_WL1F   (OFF_WHHF  + (size_t)64*32*512*2)  // wl1 f16 frags: 32nb*32kb*512
#define OFF_WL2F   (OFF_WL1F  + (size_t)32*32*512*2)
#define OFF_HHIST  (OFF_WL2F  + (size_t)32*32*512*2)  // [T][1024] f32

// dynamic LDS layout for the scan kernel (w_hh lives in VGPRs)
#define L_W1T_B   ((size_t)ODE_H*HID*2)               // 102400
#define L_W2HT_B  ((size_t)COLS*64*4)                 // 16384
#define L_WIHT_B  ((size_t)COLS*64*4)                 // 16384
#define SCAN_DYN  (L_W1T_B + L_W2HT_B + L_WIHT_B)     // 135168 B

// K-index inside a 16x32 f16 fragment for (lane-half hi, element e).
__device__ __forceinline__ int frag_klocal(int hi, int e) {
  return (e < 8) ? (hi * 8 + e) : (16 + hi * 8 + (e - 8));
}

// ---------------------------------------------------------------------------
// Pack kernels
// ---------------------------------------------------------------------------
__global__ void k_init_bar(int* bar) {
  int i = blockIdx.x * blockDim.x + threadIdx.x;
  if (i < T_STEPS) bar[i] = 0;
}

// W2H^T[n][j] = sum_m w2[j][m]*w_hh[m][n];  b2h[n]; bsum[n]
__global__ void k_pack_aux(const float* __restrict__ w2, const float* __restrict__ w_hh,
                           const float* __restrict__ b2, const float* __restrict__ b_ih,
                           const float* __restrict__ b_hh,
                           float* __restrict__ w2ht, float* __restrict__ b2h,
                           float* __restrict__ bsum) {
  int n = blockIdx.x, j = threadIdx.x;
  if (j < ODE_H) {
    float acc = 0.f;
    for (int m = 0; m < HID; ++m) acc += w2[j * HID + m] * w_hh[(size_t)m * HID + n];
    w2ht[n * 64 + j] = acc;
  } else if (j == ODE_H) {
    float acc = 0.f;
    for (int m = 0; m < HID; ++m) acc += b2[m] * w_hh[(size_t)m * HID + n];
    b2h[n] = acc;
  } else if (j == ODE_H + 1) {
    bsum[n] = b_ih[n] + b_hh[n];
  }
}

__global__ void k_pack_w1t(const float* __restrict__ w1, _Float16* __restrict__ w1t) {
  int idx = blockIdx.x * blockDim.x + threadIdx.x;
  if (idx < ODE_H * HID) {
    int j = idx >> 10, k = idx & 1023;
    w1t[idx] = (_Float16)w1[k * ODE_H + j];
  }
}

__global__ void k_pack_wiht(const float* __restrict__ w_ih, float* __restrict__ wiht) {
  int idx = blockIdx.x * blockDim.x + threadIdx.x;
  if (idx < HID * IN_DIM) {
    int n = idx >> 6, i = idx & 63;
    wiht[idx] = w_ih[i * HID + n];
  }
}

// Pack a row-major [K x N] f32 matrix into f16 WMMA-B fragments.
// frag(nb,kb) occupies 512 halves at ((nb*KB + kb)<<9), entry = lane*16 + e.
__global__ void k_pack_frags(const float* __restrict__ W, _Float16* __restrict__ F,
                             int K, int N) {
  int idx = blockIdx.x * blockDim.x + threadIdx.x;
  if (idx >= K * N) return;
  int r = idx & 511, frag = idx >> 9;
  int KB = K >> 5;
  int nb = frag / KB, kb = frag - nb * KB;
  int lane = r >> 4, e = r & 15;
  int n = nb * 16 + (lane & 15);
  int k = kb * 32 + frag_klocal(lane >> 4, e);
  F[idx] = (_Float16)W[(size_t)k * N + n];
}

// ---------------------------------------------------------------------------
// Pass 1: sequential scan. 16 persistent WGPs, one barrier per step.
//   pre[n] = x_t@w_ih[:,n] + h@w_hh[:,n] + dt*(z@W2H[:,n] + b2h[n]) + bsum[n]
//   h_new  = tanh(pre),  z = tanh(h@w1 + b1)
// w_hh B-fragments are loaded ONCE into VGPRs (16 frags x 8 VGPR per wave)
// and stay resident for all 16384 steps. K accumulation runs as two
// independent WMMA chains to break the serial C dependency.
// ---------------------------------------------------------------------------
__global__ void __launch_bounds__(TPB, 1)
k_scan(const float* __restrict__ x, const float* __restrict__ dt,
       const float* __restrict__ b1g,
       const _Float16* __restrict__ w1t_g, const float* __restrict__ w2ht_g,
       const float* __restrict__ b2h_g, const float* __restrict__ bsum_g,
       const _Float16* __restrict__ whhf_g, const float* __restrict__ wiht_g,
       float* __restrict__ h_hist, int* __restrict__ bar) {
  extern __shared__ char dyn[];
  _Float16* w1t  = (_Float16*)dyn;                         // [50][1024] f16
  float*    w2ht = (float*)(dyn + L_W1T_B);                // [64][64]
  float*    wiht = w2ht + COLS * 64;                       // [64][64]

  __shared__ float    sh_h[HID];
  __shared__ _Float16 sh_hf[HID + 32];   // +32 halves: zero tail for A-lanes
  __shared__ float    sh_z[64], sh_zacc[64], sh_pre[64];
  __shared__ float    sh_b1[64], sh_bsum[64], sh_b2h[64];

  const int tid = threadIdx.x, g = blockIdx.x;
  const int lane = tid & 31, wave = tid >> 5;
  const int m = lane & 15, hi = lane >> 4;
  const int nbl = wave >> 1, kh = wave & 1;   // nblock 0..3, K-half 0..1

  // ---- one-time preload of LDS-resident weights ----
  {
    const unsigned* s = (const unsigned*)w1t_g;
    unsigned* d = (unsigned*)w1t;
    for (int i = tid; i < (int)(L_W1T_B / 4); i += TPB) d[i] = s[i];
  }
  for (int i = tid; i < COLS * 64; i += TPB) w2ht[i] = w2ht_g[(size_t)g * COLS * 64 + i];
  for (int i = tid; i < COLS * 64; i += TPB) wiht[i] = wiht_g[(size_t)g * COLS * 64 + i];
  if (tid < 64) {
    sh_b1[tid]   = (tid < ODE_H) ? b1g[tid] : 0.f;
    sh_bsum[tid] = bsum_g[g * COLS + tid];
    sh_b2h[tid]  = b2h_g[g * COLS + tid];
  }
  for (int i = tid; i < HID + 32; i += TPB) {
    if (i < HID) sh_h[i] = 0.f;
    sh_hf[i] = (_Float16)0.f;              // includes zero tail [HID, HID+32)
  }

  // ---- pin this wave's 16 w_hh B-fragments in VGPRs for the whole scan ----
  v16h breg[16];
#pragma unroll
  for (int i = 0; i < 16; ++i) {
    int kb = kh * 16 + i;
    breg[i] = *(const v16h*)(whhf_g +
        (((size_t)(g * 4 + nbl) * KB_HID + kb) << 9) + lane * 16);
  }
  __syncthreads();

  // A-fragment source offset: lanes with m==0 read h; all others read zeros.
  // (A row 0 = h; rows 1..15 = 0; D row 0 carries the mat-vec result.)
  const int aoff_zero = HID;               // sh_hf[HID..HID+23] are zeros

  for (int t = 0; t < T_STEPS; ++t) {
    if (tid < 64) { sh_zacc[tid] = 0.f; sh_pre[tid] = 0.f; }
    __syncthreads();

    // z = tanh(h @ w1 + b1): 200 threads, 4-way K split + LDS reduce
    if (tid < 4 * ODE_H) {
      int j = tid >> 2, q = tid & 3;
      float acc = 0.f;
      const _Float16* wr = w1t + j * HID + q * 256;
      const float*    hr = sh_h + q * 256;
#pragma unroll 8
      for (int k = 0; k < 256; ++k) acc += hr[k] * (float)wr[k];
      atomicAdd(&sh_zacc[j], acc);
    }
    __syncthreads();
    if (tid < ODE_H) sh_z[tid] = tanhf(sh_zacc[tid] + sh_b1[tid]);

    // h @ w_hh slice via WMMA, B operands in registers, dual chains
    {
      v8f c0, c1;
#pragma unroll
      for (int r = 0; r < 8; ++r) { c0[r] = 0.f; c1[r] = 0.f; }
#pragma unroll
      for (int i = 0; i < 8; ++i) {
        int kb0 = kh * 16 + 2 * i;
        int kb1 = kb0 + 1;
        int off0 = (m == 0) ? (kb0 * 32 + hi * 8) : aoff_zero;
        int off1 = (m == 0) ? (kb1 * 32 + hi * 8) : aoff_zero;
        const _Float16* p0 = sh_hf + off0;
        const _Float16* p1 = sh_hf + off1;
        v16h a0, a1;                 // distinct register sets -> pair the loads
#pragma unroll
        for (int e = 0; e < 8; ++e) a0[e] = p0[e];
#pragma unroll
        for (int e = 0; e < 8; ++e) a0[e + 8] = p0[16 + e];
#pragma unroll
        for (int e = 0; e < 8; ++e) a1[e] = p1[e];
#pragma unroll
        for (int e = 0; e < 8; ++e) a1[e + 8] = p1[16 + e];
        c0 = __builtin_amdgcn_wmma_f32_16x16x32_f16(false, a0, false, breg[2 * i],
                                                    (short)0, c0, false, false);
        c1 = __builtin_amdgcn_wmma_f32_16x16x32_f16(false, a1, false, breg[2 * i + 1],
                                                    (short)0, c1, false, false);
      }
      // only D row 0 is meaningful: merge the two chains and reduce
      if (lane < 16) atomicAdd(&sh_pre[nbl * 16 + lane], c0[0] + c1[0]);
    }
    __syncthreads();

    // combine + tanh + store this WGP's 64 columns of h_t
    if (tid < COLS) {
      float acc = sh_pre[tid] + sh_bsum[tid];
      const float* xr = x + (size_t)t * IN_DIM;
      const float* wi = wiht + tid * 64;
#pragma unroll 8
      for (int i = 0; i < IN_DIM; ++i) acc += xr[i] * wi[i];
      float od = sh_b2h[tid];
      const float* w2r = w2ht + tid * 64;
#pragma unroll 5
      for (int j = 0; j < ODE_H; ++j) od += sh_z[j] * w2r[j];
      acc += dt[t] * od;
      h_hist[(size_t)t * HID + g * COLS + tid] = tanhf(acc);
    }
    __threadfence();
    __syncthreads();

    // per-step device barrier (fresh counter each step; zeroed by k_init_bar)
    if (tid == 0) {
      __hip_atomic_fetch_add(&bar[t], 1, __ATOMIC_RELEASE, __HIP_MEMORY_SCOPE_AGENT);
      while (__hip_atomic_load(&bar[t], __ATOMIC_ACQUIRE, __HIP_MEMORY_SCOPE_AGENT) < NWG)
        __builtin_amdgcn_s_sleep(1);
    }
    __syncthreads();

    // all-gather full h_t (4KB from L2)
#pragma unroll
    for (int r = 0; r < HID / TPB; ++r) {
      int i = r * TPB + tid;
      float v = h_hist[(size_t)t * HID + i];
      sh_h[i] = v;
      sh_hf[i] = (_Float16)v;
    }
    __syncthreads();
  }
}

// ---------------------------------------------------------------------------
// Pass 2: heads. Block = 16 timesteps; 8 waves x 8 jobs cover
// {mu,sigma} x 32 N-blocks of the 1024x512 GEMMs, K accumulated by WMMA.
// ---------------------------------------------------------------------------
__global__ void __launch_bounds__(TPB, 1)
k_heads(const float* __restrict__ h_hist,
        const _Float16* __restrict__ wl1f, const _Float16* __restrict__ wl2f,
        const float* __restrict__ bl1, const float* __restrict__ bl2,
        const float* __restrict__ wmu, const float* __restrict__ bmu,
        const float* __restrict__ wps, const float* __restrict__ bps,
        float* __restrict__ out) {
  __shared__ _Float16 sh_ht[16 * HID];
  __shared__ float    acc[2][16];

  const int tid = threadIdx.x;
  const int t0 = blockIdx.x * 16;
  for (int i = tid; i < 16 * HID; i += TPB)
    sh_ht[i] = (_Float16)h_hist[(size_t)t0 * HID + i];
  if (tid < 32) acc[tid >> 4][tid & 15] = 0.f;
  __syncthreads();

  const int lane = tid & 31, wave = tid >> 5;
  const int m = lane & 15, hi = lane >> 4;

  v8f c[8];
#pragma unroll
  for (int q = 0; q < 8; ++q)
#pragma unroll
    for (int r = 0; r < 8; ++r) c[q][r] = 0.f;

  for (int kb = 0; kb < KB_HID; ++kb) {
    // A fragment: two contiguous 8-half runs -> 2x ds_load_b128
    const _Float16* p = sh_ht + m * HID + kb * 32 + hi * 8;
    v16h a;
#pragma unroll
    for (int e = 0; e < 8; ++e) a[e] = p[e];
#pragma unroll
    for (int e = 0; e < 8; ++e) a[e + 8] = p[16 + e];
#pragma unroll
    for (int q = 0; q < 8; ++q) {
      int job = wave * 8 + q;            // 0..63
      int head = job >> 5, nb = job & 31;
      const _Float16* src = head ? wl2f : wl1f;
      v16h b = *(const v16h*)(src + ((((size_t)nb * KB_HID + kb) << 9) + lane * 16));
      c[q] = __builtin_amdgcn_wmma_f32_16x16x32_f16(false, a, false, b,
                                                    (short)0, c[q], false, false);
    }
  }

  // relu + collapse 512 -> 1 with the head vector; cross-lane reduce
#pragma unroll
  for (int q = 0; q < 8; ++q) {
    int job = wave * 8 + q;
    int head = job >> 5, nb = job & 31;
    int n = nb * 16 + m;
    float bias = head ? bl2[n] : bl1[n];
    float wv   = head ? wps[n] : wmu[n];
#pragma unroll
    for (int r = 0; r < 8; ++r) {
      int mm = hi ? (8 + r) : r;
      float u = c[q][r] + bias;
      u = u > 0.f ? u : 0.f;
      float v = u * wv;
#pragma unroll
      for (int s = 1; s < 16; s <<= 1) v += __shfl_xor(v, s, 32);
      if (m == 0) atomicAdd(&acc[head][mm], v);
    }
  }
  __syncthreads();

  if (tid < 16) {
    out[t0 + tid] = acc[0][tid] + bmu[0];
  } else if (tid < 32) {
    float v = acc[1][tid - 16] + bps[0];
    float sp = fmaxf(v, 0.f) + log1pf(expf(-fabsf(v)));   // stable softplus
    out[T_STEPS + t0 + (tid - 16)] = sp;
  }
}

// ---------------------------------------------------------------------------
extern "C" void kernel_launch(void* const* d_in, const int* in_sizes, int n_in,
                              void* d_out, int out_size, void* d_ws, size_t ws_size,
                              hipStream_t stream) {
  (void)in_sizes; (void)n_in; (void)out_size; (void)ws_size;
  const float* x    = (const float*)d_in[0];
  const float* dt   = (const float*)d_in[1];
  const float* w1   = (const float*)d_in[2];
  const float* b1   = (const float*)d_in[3];
  const float* w2   = (const float*)d_in[4];
  const float* b2   = (const float*)d_in[5];
  const float* w_ih = (const float*)d_in[6];
  const float* w_hh = (const float*)d_in[7];
  const float* b_ih = (const float*)d_in[8];
  const float* b_hh = (const float*)d_in[9];
  const float* wl1  = (const float*)d_in[10];
  const float* bl1  = (const float*)d_in[11];
  const float* wmu  = (const float*)d_in[12];
  const float* bmu  = (const float*)d_in[13];
  const float* wl2  = (const float*)d_in[14];
  const float* bl2  = (const float*)d_in[15];
  const float* wps  = (const float*)d_in[16];
  const float* bps  = (const float*)d_in[17];
  float* out = (float*)d_out;

  char* ws = (char*)d_ws;
  int*      barp  = (int*)(ws + OFF_BAR);
  float*    w2ht  = (float*)(ws + OFF_W2HT);
  float*    b2h   = (float*)(ws + OFF_B2H);
  float*    bsum  = (float*)(ws + OFF_BSUM);
  _Float16* w1t   = (_Float16*)(ws + OFF_W1T);
  float*    wiht  = (float*)(ws + OFF_WIHT);
  _Float16* whhf  = (_Float16*)(ws + OFF_WHHF);
  _Float16* wl1f  = (_Float16*)(ws + OFF_WL1F);
  _Float16* wl2f  = (_Float16*)(ws + OFF_WL2F);
  float*    hhist = (float*)(ws + OFF_HHIST);

  // Pass 0: barrier init + weight packing / algebraic fold
  k_init_bar<<<T_STEPS / 256, 256, 0, stream>>>(barp);
  k_pack_aux<<<HID, 64, 0, stream>>>(w2, w_hh, b2, b_ih, b_hh, w2ht, b2h, bsum);
  k_pack_w1t<<<(ODE_H * HID + 255) / 256, 256, 0, stream>>>(w1, w1t);
  k_pack_wiht<<<(HID * IN_DIM + 255) / 256, 256, 0, stream>>>(w_ih, wiht);
  k_pack_frags<<<(HID * HID + 255) / 256, 256, 0, stream>>>(w_hh, whhf, HID, HID);
  k_pack_frags<<<(HID * HHALF + 255) / 256, 256, 0, stream>>>(wl1, wl1f, HID, HHALF);
  k_pack_frags<<<(HID * HHALF + 255) / 256, 256, 0, stream>>>(wl2, wl2f, HID, HHALF);

  // Pass 1: sequential scan (135 KB dynamic LDS; w_hh pinned in VGPRs)
  hipFuncSetAttribute(reinterpret_cast<const void*>(k_scan),
                      hipFuncAttributeMaxDynamicSharedMemorySize, (int)SCAN_DYN);
  k_scan<<<NWG, TPB, SCAN_DYN, stream>>>(x, dt, b1, w1t, w2ht, b2h, bsum,
                                         whhf, wiht, hhist, barp);

  // Pass 2: parallel WMMA heads
  k_heads<<<T_STEPS / 16, TPB, 0, stream>>>(hhist, wl1f, wl2f, bl1, bl2,
                                            wmu, bmu, wps, bps, out);
}
